// GraphEvolution_39900246180397
// MI455X (gfx1250) — matrix operand order repaired and checked by reading, not compiled, not run on hardware
//
#include <hip/hip_runtime.h>
#include <hip/hip_bf16.h>
#include <math.h>

typedef __attribute__((ext_vector_type(16))) _Float16 v16h;
typedef __attribute__((ext_vector_type(4)))  _Float16 v4h;
typedef __attribute__((ext_vector_type(8)))  float    v8f;

#define NTOT  16384
#define NEDGE 262144
#define HC    512
#define HID   64
#define HGAT  8

// ---------- helpers ----------
__device__ __forceinline__ float gelu_exact(float x) {
    return 0.5f * x * (1.0f + erff(x * 0.70710678118654752f));
}
// order-preserving float<->uint map for atomicMax on unsigned
__device__ __forceinline__ unsigned fenc(float f) {
    unsigned u = __float_as_uint(f);
    return (u & 0x80000000u) ? ~u : (u | 0x80000000u);
}
__device__ __forceinline__ float fdec(unsigned u) {
    u = (u & 0x80000000u) ? (u & 0x7FFFFFFFu) : ~u;
    return __uint_as_float(u);
}

// f32 -> f16 streaming conversion (float4 in, 4x f16 out)
__global__ void cvt_f16_kernel(const float* __restrict__ src,
                               _Float16* __restrict__ dst, int n4) {
    int tid = blockIdx.x * blockDim.x + threadIdx.x;
    if (tid >= n4) return;
    float4 v = ((const float4*)src)[tid];
    v4h h;
    h[0] = (_Float16)v.x; h[1] = (_Float16)v.y;
    h[2] = (_Float16)v.z; h[3] = (_Float16)v.w;
    ((v4h*)dst)[tid] = h;
}

// ---------- WMMA GEMM:  C[M,N] = act( A[M,K] @ W[N,K]^T + bias[N] ) ----------
// A, W are f16. block = 128 threads (4 waves); block tile = 16 rows x 64 cols.
// grid = (N/64, M/16). K must be a multiple of 64. act: 0 = none, 2 = relu.
// A tile (16x64 f16, 2KB) is staged into LDS with per-lane async DMA
// (global_load_async_to_lds_b128, tracked on ASYNCcnt), then consumed by two
// back-to-back v_wmma_f32_16x16x32_f16 issues per k-step.
__global__ void wmma_gemm_kernel(const _Float16* __restrict__ A,
                                 const _Float16* __restrict__ W,
                                 const float* __restrict__ bias,
                                 float* __restrict__ C,
                                 int M, int N, int K, int act) {
    __shared__ __align__(16) _Float16 As[16 * 64];
    const int tid  = threadIdx.x;
    const int lane = tid & 31;
    const int wave = tid >> 5;                 // 0..3 -> n sub-tile
    const int m0   = blockIdx.y * 16;
    const int col  = blockIdx.x * 64 + wave * 16 + (lane & 15);
    const int hi   = lane >> 4;                // 0/1 half of wave

    // async-staging assignment: thread t copies 16B: row = t/8, seg = (t%8)*8 halves
    const int sr = tid >> 3;
    const int sc = (tid & 7) << 3;
    const unsigned lds_dst =
        (unsigned)(reinterpret_cast<uintptr_t>(&As[sr * 64 + sc]));
    const _Float16* gsrc_row = A + (long)(m0 + sr) * K + sc;

    v8f acc;
    const float bv = bias[col];
#pragma unroll
    for (int i = 0; i < 8; ++i) acc[i] = bv;

    const int mr = lane & 15;
    for (int k0 = 0; k0 < K; k0 += 64) {
        __syncthreads();   // previous tile fully consumed before DMA overwrites it
        unsigned long long ga = (unsigned long long)(gsrc_row + k0);
        asm volatile("global_load_async_to_lds_b128 %0, %1, off"
                     :: "v"(lds_dst), "v"(ga) : "memory");
        asm volatile("s_wait_asynccnt 0x0" ::: "memory");
        __syncthreads();

        // A fragments in contiguous k order (same order as B):
        // lanes 0-15: K = 0..15; lanes 16-31: K = 16..31
        v16h af0, af1;
#pragma unroll
        for (int j = 0; j < 16; ++j) {
            int kk = (hi << 4) + j;
            af0[j] = As[mr * 64 + kk];
            af1[j] = As[mr * 64 + 32 + kk];
        }
        // B fragments: row N=col of W; lanes 0-15 hold K lo 16, lanes 16-31 hi 16
        v16h bf0, bf1;
        const _Float16* wr = W + (long)col * K + k0 + (hi << 4);
#pragma unroll
        for (int j = 0; j < 16; ++j) { bf0[j] = wr[j]; bf1[j] = wr[32 + j]; }

        acc = __builtin_amdgcn_wmma_f32_16x16x32_f16(false, af0, false, bf0,
                                                     (short)0, acc, false, false);
        acc = __builtin_amdgcn_wmma_f32_16x16x32_f16(false, af1, false, bf1,
                                                     (short)0, acc, false, false);
    }

    // C/D layout: VGPR v -> M = m0 + v + 8*hi, N = col
#pragma unroll
    for (int v = 0; v < 8; ++v) {
        int row = m0 + v + (hi << 3);
        float x = acc[v];
        if (act == 2) x = fmaxf(x, 0.0f);
        C[(long)row * N + col] = x;
    }
}

// ---------- small kernels ----------
__global__ void build_y_kernel(const float* __restrict__ x, const float* __restrict__ params,
                               float* __restrict__ Y) {
    int tid = blockIdx.x * blockDim.x + threadIdx.x;
    if (tid >= NTOT * 8) return;
    int i = tid >> 3, c = tid & 7;
    Y[tid] = (c < 6) ? x[i * 6 + c] : params[c - 6];
}

// K=8 input projection: X[i,j] = b[j] + sum_k Y[i,k]*W[j,k]
__global__ void lin8_kernel(const float* __restrict__ Y, const float* __restrict__ W,
                            const float* __restrict__ b, float* __restrict__ X) {
    int tid = blockIdx.x * blockDim.x + threadIdx.x;
    if (tid >= NTOT * HC) return;
    int i = tid >> 9, j = tid & 511;
    const float* y = Y + i * 8;
    const float* w = W + j * 8;
    float s = b[j];
#pragma unroll
    for (int k = 0; k < 8; ++k) s += y[k] * w[k];
    X[tid] = s;
}

// per (edge, head): score = sum_c lrelu(xl[src]+xr[dst](+ea*We)) * att ; segment max
__global__ void edge_score_kernel(const float* __restrict__ XL, const float* __restrict__ XR,
                                  const int* __restrict__ ei, const float* __restrict__ att,
                                  const float* __restrict__ We, const float* __restrict__ ea,
                                  float* __restrict__ araw, unsigned* __restrict__ amax) {
    int tid = blockIdx.x * blockDim.x + threadIdx.x;
    if (tid >= NEDGE * HGAT) return;
    int e = tid >> 3, h = tid & 7;
    int s = ei[e], d = ei[NEDGE + e];
    const float4* pl = (const float4*)(XL + (long)s * HC + h * HID);
    const float4* pr = (const float4*)(XR + (long)d * HC + h * HID);
    const float4* pa = (const float4*)(att + h * HID);
    const float4* pw = We ? (const float4*)(We + h * HID) : nullptr;
    float eaf = ea ? ea[e] : 0.0f;
    float sum = 0.0f;
#pragma unroll 4
    for (int q = 0; q < 16; ++q) {
        float4 l = pl[q], r = pr[q], a = pa[q];
        float mx = l.x + r.x, my = l.y + r.y, mz = l.z + r.z, mw = l.w + r.w;
        if (pw) { float4 w = pw[q]; mx += eaf * w.x; my += eaf * w.y; mz += eaf * w.z; mw += eaf * w.w; }
        mx = mx > 0.f ? mx : 0.2f * mx;  my = my > 0.f ? my : 0.2f * my;
        mz = mz > 0.f ? mz : 0.2f * mz;  mw = mw > 0.f ? mw : 0.2f * mw;
        sum += mx * a.x + my * a.y + mz * a.z + mw * a.w;
    }
    araw[tid] = sum;
    atomicMax(&amax[d * HGAT + h], fenc(sum));
}

__global__ void edge_exp_kernel(const int* __restrict__ ei, float* __restrict__ araw,
                                const unsigned* __restrict__ amax, float* __restrict__ den) {
    int tid = blockIdx.x * blockDim.x + threadIdx.x;
    if (tid >= NEDGE * HGAT) return;
    int e = tid >> 3, h = tid & 7;
    int d = ei[NEDGE + e];
    float v = expf(araw[tid] - fdec(amax[d * HGAT + h]));
    araw[tid] = v;
    atomicAdd(&den[d * HGAT + h], v);
}

__global__ void edge_aggr_kernel(const float* __restrict__ XL, const int* __restrict__ ei,
                                 const float* __restrict__ araw, const float* __restrict__ den,
                                 float* __restrict__ out) {
    int tid = blockIdx.x * blockDim.x + threadIdx.x;
    if (tid >= NEDGE * HGAT) return;
    int e = tid >> 3, h = tid & 7;
    int s = ei[e], d = ei[NEDGE + e];
    float w = araw[tid] / (den[d * HGAT + h] + 1e-16f);
    const float4* pl = (const float4*)(XL + (long)s * HC + h * HID);
    float* po = out + (long)d * HC + h * HID;
#pragma unroll 4
    for (int q = 0; q < 16; ++q) {
        float4 l = pl[q];
        atomicAdd(po + q * 4 + 0, l.x * w);
        atomicAdd(po + q * 4 + 1, l.y * w);
        atomicAdd(po + q * 4 + 2, l.z * w);
        atomicAdd(po + q * 4 + 3, l.w * w);
    }
}

__global__ void node_concat_kernel(const float* __restrict__ out, const float* __restrict__ b,
                                   float* __restrict__ H) {
    int tid = blockIdx.x * blockDim.x + threadIdx.x;
    if (tid >= NTOT * HC) return;
    H[tid] = gelu_exact(out[tid] + b[tid & 511]);
}

__global__ void node_mean_kernel(const float* __restrict__ out, const float* __restrict__ b,
                                 float* __restrict__ M) {
    int tid = blockIdx.x * blockDim.x + threadIdx.x;
    if (tid >= NTOT * HID) return;
    int i = tid >> 6, c = tid & 63;
    float s = 0.0f;
#pragma unroll
    for (int h = 0; h < HGAT; ++h) s += out[(long)i * HC + h * HID + c];
    M[tid] = gelu_exact(s * 0.125f + b[c]);
}

// Out = LayerNorm(X + R) over last dim (64). One wave per node, 2 channels/lane.
__global__ void add_ln_kernel(const float* __restrict__ X, const float* __restrict__ R,
                              const float* __restrict__ g, const float* __restrict__ b,
                              float* __restrict__ Out) {
    int wid  = (blockIdx.x * blockDim.x + threadIdx.x) >> 5;
    int lane = threadIdx.x & 31;
    if (wid >= NTOT) return;
    const long base = (long)wid * HID;
    float v0 = X[base + lane]      + R[base + lane];
    float v1 = X[base + lane + 32] + R[base + lane + 32];
    float s  = v0 + v1, s2 = v0 * v0 + v1 * v1;
#pragma unroll
    for (int off = 16; off > 0; off >>= 1) {
        s  += __shfl_xor(s,  off, 32);
        s2 += __shfl_xor(s2, off, 32);
    }
    float mu  = s * (1.0f / 64.0f);
    float var = s2 * (1.0f / 64.0f) - mu * mu;
    float inv = rsqrtf(var + 1e-5f);
    Out[base + lane]      = (v0 - mu) * inv * g[lane]      + b[lane];
    Out[base + lane + 32] = (v1 - mu) * inv * g[lane + 32] + b[lane + 32];
}

__global__ void head_kernel(const float* __restrict__ T, const float* __restrict__ Wd,
                            const float* __restrict__ bd, float* __restrict__ out) {
    int i = blockIdx.x * blockDim.x + threadIdx.x;
    if (i >= NTOT) return;
    float tv[HID];
#pragma unroll
    for (int c = 0; c < HID; ++c) tv[c] = tanhf(T[(long)i * HID + c]);
#pragma unroll
    for (int o = 0; o < 4; ++o) {
        float s = bd[o];
        for (int c = 0; c < HID; ++c) s += tv[c] * Wd[o * HID + c];
        out[i * 4 + o] = s;
    }
}

// ---------- host ----------
extern "C" void kernel_launch(void* const* d_in, const int* in_sizes, int n_in,
                              void* d_out, int out_size, void* d_ws, size_t ws_size,
                              hipStream_t stream) {
    const float* x      = (const float*)d_in[0];
    const int*   ei     = (const int*)d_in[1];
    const float* ea     = (const float*)d_in[2];
    const float* params = (const float*)d_in[3];
    int wi = 4;
    auto nf = [&]() { return (const float*)d_in[wi++]; };
    const float *Wl1=nf(), *bl1=nf(), *Wr1=nf(), *br1=nf(), *We1=nf(), *att1=nf(), *b1=nf();
    const float *Wl2=nf(), *bl2=nf(), *Wr2=nf(), *br2=nf(), *att2=nf(), *b2=nf();
    const float *Wl3=nf(), *bl3=nf(), *Wr3=nf(), *br3=nf(), *att3=nf(), *b3=nf();
    struct DecW { const float *saw,*sab,*sao,*saob,*caw,*cab,*cao,*caob,*f1,*f1b,*f2,*f2b,
                              *n1g,*n1b,*n2g,*n2b,*n3g,*n3b; } dw[2];
    for (int i = 0; i < 2; ++i) {
        dw[i].saw=nf(); dw[i].sab=nf(); dw[i].sao=nf(); dw[i].saob=nf();
        dw[i].caw=nf(); dw[i].cab=nf(); dw[i].cao=nf(); dw[i].caob=nf();
        dw[i].f1=nf();  dw[i].f1b=nf(); dw[i].f2=nf();  dw[i].f2b=nf();
        dw[i].n1g=nf(); dw[i].n1b=nf(); dw[i].n2g=nf(); dw[i].n2b=nf();
        dw[i].n3g=nf(); dw[i].n3b=nf();
    }
    const float *Wd=nf(), *bd=nf();

    // workspace layout (float slots)
    float* w = (float*)d_ws;
    const size_t NC = (size_t)NTOT * HC;      // 8,388,608
    float*    H    = w;            w += NC;
    float*    XL   = w;            w += NC;
    float*    XR   = w;            w += NC;
    float*    OUT  = w;            w += NC;
    float*    ARAW = w;            w += (size_t)NEDGE * HGAT;
    unsigned* AMAX = (unsigned*)w; w += (size_t)NTOT * HGAT;
    float*    DEN  = w;            w += (size_t)NTOT * HGAT;
    float*    Y0   = w;            w += (size_t)NTOT * 8;
    float*    MEM  = w;            w += (size_t)NTOT * HID;
    float*    T    = w;            w += (size_t)NTOT * HID;
    float*    V    = w;            w += (size_t)NTOT * 128;
    float*    O    = w;            w += (size_t)NTOT * HID;
    _Float16* A16  = (_Float16*)w; w += NC / 2;           // f16 activations
    _Float16* W16  = (_Float16*)w; w += (HC * HC) / 2;    // f16 weights
    if ((size_t)((char*)w - (char*)d_ws) > ws_size) return;  // not enough scratch

    const int TPB = 256;
    const int EB  = (NEDGE * HGAT + TPB - 1) / TPB;
    const int NB  = ((int)NC + TPB - 1) / TPB;
    dim3 blk128(128);

    auto cvt = [&](const float* src, _Float16* dst, int n) {
        cvt_f16_kernel<<<(n / 4 + TPB - 1) / TPB, TPB, 0, stream>>>(src, dst, n / 4);
    };
    auto run_edges = [&](const float* att, const float* We, const float* eav,
                         const float* bias, bool concat) {
        hipMemsetAsync(AMAX, 0, (size_t)NTOT * HGAT * sizeof(unsigned), stream);
        hipMemsetAsync(DEN,  0, (size_t)NTOT * HGAT * sizeof(float),    stream);
        hipMemsetAsync(OUT,  0, NC * sizeof(float),                     stream);
        edge_score_kernel<<<EB, TPB, 0, stream>>>(XL, XR, ei, att, We, eav, ARAW, AMAX);
        edge_exp_kernel<<<EB, TPB, 0, stream>>>(ei, ARAW, AMAX, DEN);
        edge_aggr_kernel<<<EB, TPB, 0, stream>>>(XL, ei, ARAW, DEN, OUT);
        if (concat)
            node_concat_kernel<<<NB, TPB, 0, stream>>>(OUT, bias, H);
        else
            node_mean_kernel<<<(NTOT * HID + TPB - 1) / TPB, TPB, 0, stream>>>(OUT, bias, MEM);
    };
    // A16 must already hold the f16 copy of A ([NTOT,K]); converts W then runs GEMM
    auto gemm = [&](const float* Wm, const float* bias, float* C, int N, int K, int act) {
        cvt(Wm, W16, N * K);
        dim3 grid(N / 64, NTOT / 16);
        wmma_gemm_kernel<<<grid, blk128, 0, stream>>>(A16, W16, bias, C, NTOT, N, K, act);
    };

    // ---- GAT layer 1 (K=8 input proj, edge features) ----
    build_y_kernel<<<(NTOT * 8 + TPB - 1) / TPB, TPB, 0, stream>>>(x, params, Y0);
    lin8_kernel<<<NB, TPB, 0, stream>>>(Y0, Wl1, bl1, XL);
    lin8_kernel<<<NB, TPB, 0, stream>>>(Y0, Wr1, br1, XR);
    run_edges(att1, We1, ea, b1, true);

    // ---- GAT layer 2 (WMMA 512x512 GEMMs) ----
    cvt(H, A16, (int)NC);
    gemm(Wl2, bl2, XL, HC, HC, 0);
    gemm(Wr2, br2, XR, HC, HC, 0);
    run_edges(att2, nullptr, nullptr, b2, true);

    // ---- GAT layer 3 (mean over heads) ----
    cvt(H, A16, (int)NC);
    gemm(Wl3, bl3, XL, HC, HC, 0);
    gemm(Wr3, br3, XR, HC, HC, 0);
    run_edges(att3, nullptr, nullptr, b3, false);

    // ---- transformer decoder: seq_len==1 => MHA reduces to Wo(Wv(kv)+bv)+bo ----
    hipMemcpyAsync(T, MEM, (size_t)NTOT * HID * sizeof(float),
                   hipMemcpyDeviceToDevice, stream);
    const int LNB = (NTOT * 32 + TPB - 1) / TPB;  // one wave (32 threads) per node
    for (int i = 0; i < 2; ++i) {
        const DecW& d = dw[i];
        cvt(T, A16, NTOT * HID);
        gemm(d.saw + 2 * HID * HID, d.sab + 2 * HID, V, HID, HID, 0); // v-proj (self)
        cvt(V, A16, NTOT * HID);
        gemm(d.sao, d.saob, O, HID, HID, 0);                          // out-proj
        add_ln_kernel<<<LNB, TPB, 0, stream>>>(T, O, d.n1g, d.n1b, T);
        cvt(MEM, A16, NTOT * HID);
        gemm(d.caw + 2 * HID * HID, d.cab + 2 * HID, V, HID, HID, 0); // v-proj (cross)
        cvt(V, A16, NTOT * HID);
        gemm(d.cao, d.caob, O, HID, HID, 0);
        add_ln_kernel<<<LNB, TPB, 0, stream>>>(T, O, d.n2g, d.n2b, T);
        cvt(T, A16, NTOT * HID);
        gemm(d.f1, d.f1b, V, 128, HID, 2);                            // FF1 + relu
        cvt(V, A16, NTOT * 128);
        gemm(d.f2, d.f2b, O, HID, 128, 0);                            // FF2
        add_ln_kernel<<<LNB, TPB, 0, stream>>>(T, O, d.n3g, d.n3b, T);
    }

    // ---- tanh + output linear ----
    head_kernel<<<(NTOT + TPB - 1) / TPB, TPB, 0, stream>>>(T, Wd, bd, (float*)d_out);
}
